// NaViT_86397562126359
// MI455X (gfx1250) — matrix operand order repaired and checked by reading, not compiled, not run on hardware
//
#include <hip/hip_runtime.h>
#include <hip/hip_bf16.h>
#include <math.h>

#define B_   4
#define L_   1536
#define D_   768
#define H_   12
#define HD_  64
#define FF_  3072
#define NL_  2
#define NC_  1000
#define NEG_ (-1e9f)

typedef __attribute__((ext_vector_type(2)))  __bf16 v2bf;
typedef __attribute__((ext_vector_type(8)))  __bf16 v8bf;
typedef __attribute__((ext_vector_type(16))) __bf16 v16bf;
typedef __attribute__((ext_vector_type(8)))  float  v8f;

static __device__ __forceinline__ v8f zero_v8f() {
  v8f z;
#pragma unroll
  for (int i = 0; i < 8; ++i) z[i] = 0.0f;
  return z;
}

// A-fragment (16x32 bf16): lane m = lane&15; k runs [8*hi,8*hi+8) and [16+8*hi,...)
static __device__ __forceinline__ v16bf frag_ld_a(const __bf16* row, int hi) {
  v8bf lo = *(const v8bf*)(row + 8 * hi);
  v8bf hb = *(const v8bf*)(row + 16 + 8 * hi);
  return __builtin_shufflevector(lo, hb, 0,1,2,3,4,5,6,7,8,9,10,11,12,13,14,15);
}
// B-fragment (32x16 bf16): lane n = lane&15; k runs [8*hi,8*hi+8) and [16+8*hi,...)
static __device__ __forceinline__ v16bf frag_ld_b(const __bf16* row, int hi) {
  v8bf lo = *(const v8bf*)(row + 8 * hi);
  v8bf hb = *(const v8bf*)(row + 16 + 8 * hi);
  return __builtin_shufflevector(lo, hb, 0,1,2,3,4,5,6,7,8,9,10,11,12,13,14,15);
}

static __device__ __forceinline__ v8f wmma_bf16(v16bf a, v16bf b, v8f c) {
  return __builtin_amdgcn_wmma_f32_16x16x32_bf16(false, a, false, b, (short)0, c,
                                                 false, false);
}

// pack two f32 -> one bf16x2 32-bit LDS store
static __device__ __forceinline__ void st_pair(__bf16* dst, float a, float b) {
  v2bf t = { (__bf16)a, (__bf16)b };
  *(v2bf*)dst = t;
}

// -------------------- Generic bf16 WMMA GEMM: C = act(A @ W + bias) ----------
// 128x128x32 block tile, 8 waves (2x4), register double-buffered global loads.
#define BM 128
#define BN 128
#define BK 32
#define GP 8  // LDS pad (halfwords) -> row stride 40 halves = 80B (16B multiple)

__global__ __launch_bounds__(256)
void gemm_bf16_kernel(const float* __restrict__ A, const float* __restrict__ W,
                      const float* __restrict__ bias, float* __restrict__ C,
                      int M, int N, int K, int act_gelu) {
  __shared__ __align__(16) __bf16 As[BM][BK + GP];
  __shared__ __align__(16) __bf16 Bs[BN][BK + GP];

  const int tid  = threadIdx.x;
  const int lane = tid & 31;
  const int wave = tid >> 5;
  const int hi   = lane >> 4;
  const int l15  = lane & 15;
  const int wm   = wave & 1;   // 2 waves along M (64 rows each)
  const int wn   = wave >> 1;  // 4 waves along N (32 cols each)
  const int m0   = blockIdx.y * BM;
  const int n0   = blockIdx.x * BN;

  // per-thread staging coordinates
  const int ar = tid >> 3;          // A row   (+ it*32)
  const int ac = (tid & 7) * 4;     // A col
  const int bk = (tid >> 4) * 2;    // B k-row pair: k, k+1  (0..30)
  const int bc = (tid & 15) * 8;    // B col group of 8      (0..120)

  v8f acc[4][2];
#pragma unroll
  for (int i = 0; i < 4; ++i)
#pragma unroll
    for (int j = 0; j < 2; ++j) acc[i][j] = zero_v8f();

  float4 aR[4], bR[4];

  auto loadB = [&](int kbase) {
#pragma unroll
    for (int kr = 0; kr < 2; ++kr) {
#pragma unroll
      for (int cg = 0; cg < 2; ++cg) {
        int gn = n0 + bc + cg * 4;
        const float* src = W + (size_t)(kbase + bk + kr) * N + gn;
        float4 v;
        if (gn + 3 < N) {
          v = *(const float4*)src;
        } else {
          v.x = (gn + 0 < N) ? src[0] : 0.f;
          v.y = (gn + 1 < N) ? src[1] : 0.f;
          v.z = (gn + 2 < N) ? src[2] : 0.f;
          v.w = (gn + 3 < N) ? src[3] : 0.f;
        }
        bR[kr * 2 + cg] = v;
      }
    }
  };
  auto loadA = [&](int kbase) {
#pragma unroll
    for (int it = 0; it < 4; ++it) {
      int gm = m0 + ar + it * 32;
      aR[it] = make_float4(0.f, 0.f, 0.f, 0.f);
      if (gm < M) aR[it] = *(const float4*)(A + (size_t)gm * K + kbase + ac);
    }
  };

  // ---- prologue: fetch K-tile 0 into registers ----
  loadA(0);
  loadB(0);

  for (int k0 = 0; k0 < K; k0 += BK) {
    // ---- commit staged registers to LDS (f32 -> bf16) ----
#pragma unroll
    for (int it = 0; it < 4; ++it) {
      int r = ar + it * 32;
      st_pair(&As[r][ac + 0], aR[it].x, aR[it].y);
      st_pair(&As[r][ac + 2], aR[it].z, aR[it].w);
    }
    // B transposed: pack (k, k+1) pairs -> one b32 store per column
    {
      const float lo[8] = { bR[0].x, bR[0].y, bR[0].z, bR[0].w,
                            bR[1].x, bR[1].y, bR[1].z, bR[1].w };
      const float hh[8] = { bR[2].x, bR[2].y, bR[2].z, bR[2].w,
                            bR[3].x, bR[3].y, bR[3].z, bR[3].w };
#pragma unroll
      for (int j = 0; j < 8; ++j) st_pair(&Bs[bc + j][bk], lo[j], hh[j]);
    }
    __syncthreads();

    // ---- issue next-tile global loads now; latency hides under WMMA below ----
    const int kn = k0 + BK;
    if (kn < K) {
      loadA(kn);
      loadB(kn);
      if (kn + BK < K) {  // warm L2 two tiles ahead
        __builtin_prefetch(A + (size_t)(m0 + ar) * K + kn + BK + ac, 0, 1);
        __builtin_prefetch(W + (size_t)(kn + BK + bk) * N + n0 + bc, 0, 1);
      }
    }

    // ---- fragment loads + WMMA ----
    v16bf af[4], bfr[2];
#pragma unroll
    for (int ms = 0; ms < 4; ++ms)
      af[ms] = frag_ld_a(&As[wm * 64 + ms * 16 + l15][0], hi);
#pragma unroll
    for (int ns = 0; ns < 2; ++ns)
      bfr[ns] = frag_ld_b(&Bs[wn * 32 + ns * 16 + l15][0], hi);
#pragma unroll
    for (int ms = 0; ms < 4; ++ms)
#pragma unroll
      for (int ns = 0; ns < 2; ++ns)
        acc[ms][ns] = wmma_bf16(af[ms], bfr[ns], acc[ms][ns]);
    __syncthreads();
  }

  // ---- epilogue: bias + optional exact GELU ----
#pragma unroll
  for (int ms = 0; ms < 4; ++ms) {
#pragma unroll
    for (int ns = 0; ns < 2; ++ns) {
      int n = n0 + wn * 32 + ns * 16 + l15;
      float bv = 0.f;
      if (bias != nullptr && n < N) bv = bias[n];
#pragma unroll
      for (int r = 0; r < 8; ++r) {
        int m = m0 + wm * 64 + ms * 16 + hi * 8 + r;
        if (m < M && n < N) {
          float v = acc[ms][ns][r] + bv;
          if (act_gelu) v = 0.5f * v * (1.f + erff(v * 0.70710678118654752f));
          C[(size_t)m * N + n] = v;
        }
      }
    }
  }
}

// -------------------- Flash attention (per (b,h), 64-query blocks) -----------
__global__ __launch_bounds__(128)
void flash_attn_kernel(const float* __restrict__ qb, const float* __restrict__ kvb,
                       const int* __restrict__ idx, const int* __restrict__ slens,
                       float* __restrict__ ob) {
  __shared__ __align__(16) __bf16 Qs[64][72];        // [q][hd]
  __shared__ __align__(16) __bf16 Ks[64][72];        // [key][hd]
  __shared__ __align__(16) __bf16 Vs[64][72];        // [hd][key] (transposed)
  __shared__ __align__(16) __bf16 Ps[4][16][72];     // per-wave P tile [q][key]
  __shared__ int qidx[64];
  __shared__ int kidx[64];

  const int tid  = threadIdx.x;
  const int lane = tid & 31;
  const int wave = tid >> 5;
  const int hi   = lane >> 4;
  const int l15  = lane & 15;
  const int q0   = blockIdx.x * 64;
  const int bh   = blockIdx.y;
  const int bb   = bh / H_;
  const int h    = bh % H_;
  const int slen = slens[bb];
  const int qrow = wave * 16;

  // load Q block (64x64 f32 -> bf16)
#pragma unroll
  for (int it = 0; it < 8; ++it) {
    int r = (tid >> 4) + it * 8;
    int c = (tid & 15) * 4;
    float4 v = *(const float4*)(qb + ((size_t)(bb * L_ + q0 + r)) * D_ + h * HD_ + c);
    st_pair(&Qs[r][c + 0], v.x, v.y);
    st_pair(&Qs[r][c + 2], v.z, v.w);
  }
  if (tid < 64) qidx[tid] = idx[bb * L_ + q0 + tid];

  float m_run[8], l_run[8];
  v8f o_acc[4];
#pragma unroll
  for (int r = 0; r < 8; ++r) { m_run[r] = -INFINITY; l_run[r] = 0.f; }
#pragma unroll
  for (int i = 0; i < 4; ++i) o_acc[i] = zero_v8f();

  for (int k0 = 0; k0 < L_; k0 += 64) {
    __syncthreads();  // protect Ks/Vs/kidx from previous iteration readers
    // K rows (hd contiguous)
#pragma unroll
    for (int it = 0; it < 8; ++it) {
      int r = (tid >> 4) + it * 8;
      int c = (tid & 15) * 4;
      const float* kp = kvb + ((size_t)(bb * L_ + k0 + r)) * (2 * D_) + h * HD_ + c;
      float4 kv4 = *(const float4*)kp;
      st_pair(&Ks[r][c + 0], kv4.x, kv4.y);
      st_pair(&Ks[r][c + 2], kv4.z, kv4.w);
    }
    // V transposed: key pairs -> one b32 store per hd column
#pragma unroll
    for (int it = 0; it < 4; ++it) {
      int k2 = ((tid >> 4) + it * 8) * 2;   // 0..62 even
      int c  = (tid & 15) * 4;
      const float* v0p = kvb + ((size_t)(bb * L_ + k0 + k2)) * (2 * D_) + D_ + h * HD_ + c;
      const float* v1p = v0p + 2 * D_;
      float4 v0 = *(const float4*)v0p;
      float4 v1 = *(const float4*)v1p;
      st_pair(&Vs[c + 0][k2], v0.x, v1.x);
      st_pair(&Vs[c + 1][k2], v0.y, v1.y);
      st_pair(&Vs[c + 2][k2], v0.z, v1.z);
      st_pair(&Vs[c + 3][k2], v0.w, v1.w);
    }
    if (tid < 64) kidx[tid] = idx[bb * L_ + k0 + tid];
    __syncthreads();

    // S = Q @ K^T  (16 q rows x 64 keys per wave)
    v16bf aq0 = frag_ld_a(&Qs[qrow + l15][0], hi);
    v16bf aq1 = frag_ld_a(&Qs[qrow + l15][32], hi);
    v8f s[4];
#pragma unroll
    for (int nf = 0; nf < 4; ++nf) {
      v16bf b0 = frag_ld_b(&Ks[nf * 16 + l15][0], hi);
      v16bf b1 = frag_ld_b(&Ks[nf * 16 + l15][32], hi);
      v8f a = zero_v8f();
      a = wmma_bf16(aq0, b0, a);
      a = wmma_bf16(aq1, b1, a);
      s[nf] = a;
    }

    int  kid[4];
    bool kok[4];
#pragma unroll
    for (int nf = 0; nf < 4; ++nf) {
      int kl = nf * 16 + l15;
      kid[nf] = kidx[kl];
      kok[nf] = (k0 + kl) < slen;
    }

    // mask + online softmax (per q row = vgpr r + 8*hi; half-wave reductions)
#pragma unroll
    for (int r = 0; r < 8; ++r) {
      int qi = qidx[qrow + hi * 8 + r];
      float rmax = -INFINITY;
#pragma unroll
      for (int nf = 0; nf < 4; ++nf) {
        float sv = (kok[nf] && (kid[nf] == qi)) ? s[nf][r] : NEG_;
        s[nf][r] = sv;
        rmax = fmaxf(rmax, sv);
      }
#pragma unroll
      for (int off = 1; off < 16; off <<= 1)
        rmax = fmaxf(rmax, __shfl_xor(rmax, off, 32));
      float mnew  = fmaxf(m_run[r], rmax);
      float scale = __expf(m_run[r] - mnew);
      float rsum  = 0.f;
#pragma unroll
      for (int nf = 0; nf < 4; ++nf) {
        float p = __expf(s[nf][r] - mnew);
        s[nf][r] = p;
        rsum += p;
      }
#pragma unroll
      for (int off = 1; off < 16; off <<= 1) rsum += __shfl_xor(rsum, off, 32);
      l_run[r] = l_run[r] * scale + rsum;
      m_run[r] = mnew;
#pragma unroll
      for (int nf = 0; nf < 4; ++nf) o_acc[nf][r] *= scale;
    }

    // route P (D-layout) through per-wave LDS to A-layout, then O += P @ V
#pragma unroll
    for (int nf = 0; nf < 4; ++nf)
#pragma unroll
      for (int r = 0; r < 8; ++r)
        Ps[wave][hi * 8 + r][nf * 16 + l15] = (__bf16)s[nf][r];

    v16bf ap0 = frag_ld_a(&Ps[wave][l15][0], hi);
    v16bf ap1 = frag_ld_a(&Ps[wave][l15][32], hi);
#pragma unroll
    for (int nf = 0; nf < 4; ++nf) {
      v16bf bv0 = frag_ld_b(&Vs[nf * 16 + l15][0], hi);
      v16bf bv1 = frag_ld_b(&Vs[nf * 16 + l15][32], hi);
      o_acc[nf] = wmma_bf16(ap0, bv0, o_acc[nf]);
      o_acc[nf] = wmma_bf16(ap1, bv1, o_acc[nf]);
    }
  }

  // O / l -> out [B,L,H,HD]
#pragma unroll
  for (int nf = 0; nf < 4; ++nf) {
    int hd = nf * 16 + l15;
#pragma unroll
    for (int r = 0; r < 8; ++r) {
      int qg = q0 + qrow + hi * 8 + r;
      float ov = o_acc[nf][r] / fmaxf(l_run[r], 1e-20f);
      ob[((size_t)(bb * L_ + qg)) * D_ + h * HD_ + hd] = ov;
    }
  }
}

// -------------------- elementwise / small kernels ----------------------------
__global__ void add_pos_kernel(float* __restrict__ x, const int* __restrict__ pos,
                               const float* __restrict__ px, const float* __restrict__ py) {
  int row = blockIdx.x, t = threadIdx.x;
  int ix = pos[row * 2 + 0], iy = pos[row * 2 + 1];
#pragma unroll
  for (int j = 0; j < 3; ++j) {
    int d = t + j * 256;
    x[(size_t)row * D_ + d] += px[(size_t)ix * D_ + d] + py[(size_t)iy * D_ + d];
  }
}

__global__ __launch_bounds__(256)
void resid_ln_kernel(float* __restrict__ x, const float* __restrict__ rs) {
  int row = blockIdx.x, t = threadIdx.x;
  __shared__ float red[256];
  float v[3], s = 0.f;
#pragma unroll
  for (int j = 0; j < 3; ++j) {
    size_t o = (size_t)row * D_ + t + j * 256;
    float a = x[o];
    if (rs != nullptr) a += rs[o];
    v[j] = a;
    s += a;
  }
  red[t] = s; __syncthreads();
  for (int o = 128; o > 0; o >>= 1) { if (t < o) red[t] += red[t + o]; __syncthreads(); }
  float mean = red[0] * (1.f / (float)D_);
  __syncthreads();
  float s2 = 0.f;
#pragma unroll
  for (int j = 0; j < 3; ++j) { float d = v[j] - mean; s2 += d * d; }
  red[t] = s2; __syncthreads();
  for (int o = 128; o > 0; o >>= 1) { if (t < o) red[t] += red[t + o]; __syncthreads(); }
  float inv = rsqrtf(red[0] * (1.f / (float)D_) + 1e-5f);
#pragma unroll
  for (int j = 0; j < 3; ++j)
    x[(size_t)row * D_ + t + j * 256] = (v[j] - mean) * inv;
}

// F.normalize(dim=-1) * sqrt(HD) * gamma; one 64-thread block per (row, head)
__global__ void qknorm_kernel(float* __restrict__ base, int rstride,
                              const float* __restrict__ gamma) {
  int rh = blockIdx.x, row = rh / H_, h = rh % H_, t = threadIdx.x;
  float* p = base + (size_t)row * rstride + h * HD_;
  float v = p[t];
  __shared__ float red[64];
  red[t] = v * v; __syncthreads();
  for (int o = 32; o > 0; o >>= 1) { if (t < o) red[t] += red[t + o]; __syncthreads(); }
  float n  = sqrtf(red[0]);
  float sc = 8.0f / fmaxf(n, 1e-12f);  // sqrt(64) / max(n, 1e-12)
  p[t] = v * sc * gamma[h * HD_ + t];
}

// qv = pool_q @ pool_qw + pool_qb   (single block)
__global__ void poolq_kernel(const float* __restrict__ pq, const float* __restrict__ qw,
                             const float* __restrict__ qb, float* __restrict__ out) {
  int t = threadIdx.x;
#pragma unroll
  for (int j = 0; j < 3; ++j) {
    int o = t + j * 256;
    float s = qb[o];
    for (int k = 0; k < D_; ++k) s += pq[k] * qw[(size_t)k * D_ + o];
    out[o] = s;
  }
}

// single-query attention pooling per (b,h); pad-mask only; ds_add_f32 reduce
__global__ __launch_bounds__(256)
void pool_attn_kernel(const float* __restrict__ qv, const float* __restrict__ kv,
                      const int* __restrict__ slens, float* __restrict__ out) {
  int bb = blockIdx.x / H_, h = blockIdx.x % H_, t = threadIdx.x;
  const float* q = qv + h * HD_;
  int slen = slens[bb];
  __shared__ float red[256];
  __shared__ float acc[HD_];

  float sc[6];
  float lmax = -INFINITY;
#pragma unroll
  for (int i = 0; i < 6; ++i) {
    int k = t + i * 256;
    float s = NEG_;
    if (k < slen) {
      const float* kp = kv + ((size_t)(bb * L_ + k)) * (2 * D_) + h * HD_;
      s = 0.f;
#pragma unroll 8
      for (int d = 0; d < HD_; ++d) s += q[d] * kp[d];
    }
    sc[i] = s;
    lmax = fmaxf(lmax, s);
  }
  red[t] = lmax; __syncthreads();
  for (int o = 128; o > 0; o >>= 1) { if (t < o) red[t] = fmaxf(red[t], red[t + o]); __syncthreads(); }
  float gmax = red[0]; __syncthreads();

  float ls = 0.f;
#pragma unroll
  for (int i = 0; i < 6; ++i) { float p = __expf(sc[i] - gmax); sc[i] = p; ls += p; }
  red[t] = ls; __syncthreads();
  for (int o = 128; o > 0; o >>= 1) { if (t < o) red[t] += red[t + o]; __syncthreads(); }
  float gsum = red[0]; __syncthreads();

  if (t < HD_) acc[t] = 0.f;
  __syncthreads();
#pragma unroll
  for (int i = 0; i < 6; ++i) {
    int k = t + i * 256;
    if (k < slen) {
      const float* vp = kv + ((size_t)(bb * L_ + k)) * (2 * D_) + D_ + h * HD_;
      float p = sc[i];
#pragma unroll 8
      for (int d = 0; d < HD_; ++d) atomicAdd(&acc[d], p * vp[d]);
    }
  }
  __syncthreads();
  if (t < HD_) out[(size_t)bb * D_ + h * HD_ + t] = acc[t] / gsum;
}

// -------------------- driver -------------------------------------------------
extern "C" void kernel_launch(void* const* d_in, const int* in_sizes, int n_in,
                              void* d_out, int out_size, void* d_ws, size_t ws_size,
                              hipStream_t stream) {
  (void)in_sizes; (void)n_in; (void)out_size; (void)ws_size;

  const float* img     = (const float*)d_in[0];
  const int*   pos     = (const int*)d_in[1];
  const int*   idx     = (const int*)d_in[2];
  const int*   slens   = (const int*)d_in[3];
  const float* patch_w = (const float*)d_in[4];
  const float* patch_b = (const float*)d_in[5];
  const float* pos_x   = (const float*)d_in[6];
  const float* pos_y   = (const float*)d_in[7];
  const float* enc_qw  = (const float*)d_in[8];
  const float* enc_qb  = (const float*)d_in[9];
  const float* enc_kvw = (const float*)d_in[10];
  const float* enc_kvb = (const float*)d_in[11];
  const float* enc_qg  = (const float*)d_in[12];
  const float* enc_kg  = (const float*)d_in[13];
  const float* enc_ow  = (const float*)d_in[14];
  const float* enc_f1w = (const float*)d_in[15];
  const float* enc_f1b = (const float*)d_in[16];
  const float* enc_f2w = (const float*)d_in[17];
  const float* enc_f2b = (const float*)d_in[18];
  const float* pool_q  = (const float*)d_in[19];
  const float* pool_qw = (const float*)d_in[20];
  const float* pool_qb = (const float*)d_in[21];
  const float* pool_kvw= (const float*)d_in[22];
  const float* pool_kvb= (const float*)d_in[23];
  const float* pool_qg = (const float*)d_in[24];
  const float* pool_kg = (const float*)d_in[25];
  const float* pool_ow = (const float*)d_in[26];
  const float* cls_w   = (const float*)d_in[27];
  const float* cls_b   = (const float*)d_in[28];

  const int M = B_ * L_;  // 6144
  float* ws = (float*)d_ws;
  const size_t NX = (size_t)M * D_;    // 4,718,592 floats
  const size_t NH = (size_t)M * FF_;   // 18,874,368 floats
  float* x      = ws;
  float* bq     = x + NX;        // q / o-proj out / f2 out
  float* bkv    = bq + NX;       // kv (M x 2D) and FFN hidden (M x FF)
  float* batt   = bkv + NH;      // attention output [B,L,H,HD]
  float* qv     = batt + NX;     // 768 (pool query, post-proj)
  float* pooled = qv + 1024;     // B x D
  float* pproj  = pooled + 4096; // B x D

  auto gemm = [&](const float* A, const float* W, const float* bias, float* C,
                  int m, int n, int k, int gelu) {
    dim3 grid((n + BN - 1) / BN, (m + BM - 1) / BM);
    gemm_bf16_kernel<<<grid, dim3(256), 0, stream>>>(A, W, bias, C, m, n, k, gelu);
  };

  // patch embed + pos embed
  gemm(img, patch_w, patch_b, x, M, D_, D_, 0);  // TD == D == 768
  add_pos_kernel<<<M, 256, 0, stream>>>(x, pos, pos_x, pos_y);

  for (int l = 0; l < NL_; ++l) {
    gemm(x, enc_qw + (size_t)l * D_ * D_, enc_qb + (size_t)l * D_, bq, M, D_, D_, 0);
    gemm(x, enc_kvw + (size_t)l * D_ * 2 * D_, enc_kvb + (size_t)l * 2 * D_, bkv,
         M, 2 * D_, D_, 0);
    qknorm_kernel<<<M * H_, 64, 0, stream>>>(bq, D_, enc_qg + (size_t)l * D_);
    qknorm_kernel<<<M * H_, 64, 0, stream>>>(bkv, 2 * D_, enc_kg + (size_t)l * D_);
    flash_attn_kernel<<<dim3(L_ / 64, B_ * H_), 128, 0, stream>>>(bq, bkv, idx, slens, batt);
    gemm(batt, enc_ow + (size_t)l * D_ * D_, nullptr, bq, M, D_, D_, 0);
    resid_ln_kernel<<<M, 256, 0, stream>>>(x, bq);
    gemm(x, enc_f1w + (size_t)l * D_ * FF_, enc_f1b + (size_t)l * FF_, bkv, M, FF_, D_, 1);
    gemm(bkv, enc_f2w + (size_t)l * FF_ * D_, enc_f2b + (size_t)l * D_, bq, M, D_, FF_, 0);
    resid_ln_kernel<<<M, 256, 0, stream>>>(x, bq);
  }
  // final encoder LN
  resid_ln_kernel<<<M, 256, 0, stream>>>(x, nullptr);

  // attention pooling
  poolq_kernel<<<1, 256, 0, stream>>>(pool_q, pool_qw, pool_qb, qv);
  qknorm_kernel<<<H_, 64, 0, stream>>>(qv, D_, pool_qg);
  gemm(x, pool_kvw, pool_kvb, bkv, M, 2 * D_, D_, 0);
  qknorm_kernel<<<M * H_, 64, 0, stream>>>(bkv, 2 * D_, pool_kg);
  pool_attn_kernel<<<B_ * H_, 256, 0, stream>>>(qv, bkv, slens, pooled);
  gemm(pooled, pool_ow, nullptr, pproj, B_, D_, D_, 0);

  // classifier -> d_out (f32, [4,1000])
  gemm(pproj, cls_w, cls_b, (float*)d_out, B_, NC_, D_, 0);
}
